// HybridQLSTM_65481071405469
// MI455X (gfx1250) — compile-verified, matrix operand order checked by reading, not run on hardware
//
#include <hip/hip_runtime.h>

typedef __attribute__((ext_vector_type(16))) __bf16 v16bf;
typedef __attribute__((ext_vector_type(8)))  float  v8f;

constexpr int Tn = 256, Bn = 128, Dn = 1024, Hn = 1024;
constexpr int Kn = Dn + Hn; // 2048

// ---------- helpers ----------
static __device__ __forceinline__ unsigned short f2bf(float f) {
  unsigned int u = __builtin_bit_cast(unsigned int, f);
  u += 0x7fffu + ((u >> 16) & 1u);          // round-to-nearest-even
  return (unsigned short)(u >> 16);
}

struct Pack32 { uint4 lo, hi; };
static __device__ __forceinline__ v16bf frag16(uint4 lo, uint4 hi) {
  Pack32 p{lo, hi};
  return __builtin_bit_cast(v16bf, p);
}

static __device__ __forceinline__ float fast_sigmoid(float x) {
  return 1.0f / (1.0f + __expf(-x));
}
static __device__ __forceinline__ float fast_tanh(float x) {
  return 1.0f - 2.0f / (__expf(2.0f * x) + 1.0f);
}

// ---------- one-time conversion kernels ----------
__global__ void cvt_bf16_kernel(const float* __restrict__ src,
                                unsigned short* __restrict__ dst, long long n) {
  long long i = (long long)blockIdx.x * blockDim.x + threadIdx.x;
  if (i < n) dst[i] = f2bf(src[i]);
}

__global__ void cvt_weights_kernel(const float* __restrict__ Wf,
                                   const float* __restrict__ Wi,
                                   const float* __restrict__ Wg,
                                   const float* __restrict__ Wo,
                                   unsigned short* __restrict__ dst) {
  const long long per = (long long)Hn * Kn;
  long long i = (long long)blockIdx.x * blockDim.x + threadIdx.x;
  if (i >= 4 * per) return;
  int g = (int)(i / per);
  long long r = i - (long long)g * per;
  const float* W = (g == 0) ? Wf : (g == 1) ? Wi : (g == 2) ? Wg : Wo;
  dst[i] = f2bf(W[r]);
}

__global__ void init_state_kernel(unsigned short* __restrict__ H0,
                                  unsigned short* __restrict__ H1,
                                  float* __restrict__ C) {
  int i = blockIdx.x * blockDim.x + threadIdx.x; // covers Bn*Hn
  H0[i] = 0;
  H1[i] = 0;
  C[i] = 0.0f;
}

// ---------- fused per-timestep kernel ----------
// grid: (Hn/16, Bn/32), block: 128 threads = 4 waves (wave w == gate w: f,i,g,o)
// Each wave computes TWO 16x16 output tiles (b-subtiles) per gate, reusing each
// weight (B) fragment for 2 WMMAs. z-tile rows are INTERLEAVED in LDS
// (lds_row = (b&15)*2 + (b>>4)) so the two subtiles' A fragments sit exactly
// Kn*2 = 4096 bytes apart -> both reachable from one base via 16-bit DS
// immediate offsets (no per-iteration VALU address math on WMMA operand regs).
__global__ __launch_bounds__(128) void lstm_step_kernel(
    const unsigned short* __restrict__ Xt,    // [Bn, Dn]  bf16, this timestep
    const unsigned short* __restrict__ W,     // [4, Hn, Kn] bf16
    const unsigned short* __restrict__ Hprev, // [Bn, Hn]  bf16
    unsigned short* __restrict__ Hnext,       // [Bn, Hn]  bf16
    float* __restrict__ Cst,                  // [Bn, Hn]  fp32 cell state
    const float* __restrict__ bf_, const float* __restrict__ bi_,
    const float* __restrict__ bg_, const float* __restrict__ bo_,
    float* __restrict__ outT)                 // [Bn, Hn]  fp32 output slice
{
  __shared__ __align__(16) unsigned short zTile[32 * Kn]; // 128 KB, row-interleaved
  __shared__ float gAcc[4][32][16];                        // 8 KB

  const int htile = blockIdx.x * 16;
  const int btile = blockIdx.y * 32;
  const int tid   = threadIdx.x;

  // ---- stage the 32-row z tile into LDS, interleaving subtile rows ----
  {
    uint4* zT4 = (uint4*)zTile;                 // row stride = Kn/8 = 256 uint4
    for (int u = tid; u < 32 * 256; u += 128) { // 64 iters/thread
      int row = u >> 8;                         // 0..31 (b-row within tile)
      int col = u & 255;
      int ldsrow = ((row & 15) << 1) | (row >> 4); // interleave subtiles
      uint4 v;
      if (col < 128)
        v = ((const uint4*)(Xt + (size_t)(btile + row) * Dn))[col];
      else
        v = ((const uint4*)(Hprev + (size_t)(btile + row) * Hn))[col - 128];
      zT4[ldsrow * 256 + col] = v;
    }
  }
  __syncthreads();

  const int wave    = tid >> 5;   // gate index: 0=f 1=i 2=g 3=o
  const int lane    = tid & 31;
  const int halfsel = lane >> 4;  // K-half select per ISA A/B fragment layout
  const int mn      = lane & 15;  // A row (b) / B column (h) within tile

  // bias folded into accumulator init: every element of this lane's acc has
  // column h = htile + mn, so a broadcast init is exact.
  const float* bias = (wave == 0) ? bf_ : (wave == 1) ? bi_ : (wave == 2) ? bg_ : bo_;
  const float bv = bias[htile + mn];
  v8f acc0 = {bv, bv, bv, bv, bv, bv, bv, bv};
  v8f acc1 = {bv, bv, bv, bv, bv, bv, bv, bv};

  const unsigned short* Wrow  = W + ((size_t)wave * Hn + (htile + mn)) * Kn;
  const unsigned short* zRow0 = zTile + (size_t)(mn * 2) * Kn;   // subtile 0
  // subtile 1 = zRow0 + Kn shorts (4096 B, within DS immediate offset range)

  // ---- K reduction: 2 x 64 WMMAs, B fragment loaded once per pair ----
  for (int kc = 0; kc < Kn; kc += 32) {
    const int kb = kc + halfsel * 8;
    // B fragment (global, row-major weights): runs [kb,kb+8) and [kb+16,kb+24)
    uint4 b0 = *(const uint4*)(Wrow + kb);
    uint4 b1 = *(const uint4*)(Wrow + kb + 16);
    v16bf Bf = frag16(b0, b1);
    // A fragments (LDS) for both b-subtiles, constant-offset addressing
    uint4 a00 = *(const uint4*)(zRow0 + kb);
    uint4 a01 = *(const uint4*)(zRow0 + kb + 16);
    uint4 a10 = *(const uint4*)(zRow0 + kb + Kn);
    uint4 a11 = *(const uint4*)(zRow0 + kb + Kn + 16);
    acc0 = __builtin_amdgcn_wmma_f32_16x16x32_bf16(
        false, frag16(a00, a01), false, Bf, (short)0, acc0, false, false);
    acc1 = __builtin_amdgcn_wmma_f32_16x16x32_bf16(
        false, frag16(a10, a11), false, Bf, (short)0, acc1, false, false);
  }

  // ---- stage accumulators: VGPR r -> row m = 8*halfsel + r, col n = mn ----
  #pragma unroll
  for (int r = 0; r < 8; ++r) {
    gAcc[wave][halfsel * 8 + r][mn]      = acc0[r];
    gAcc[wave][16 + halfsel * 8 + r][mn] = acc1[r];
  }
  __syncthreads();

  // ---- fused LSTM cell update over the 32x16 (b,h) patch ----
  for (int e = tid; e < 512; e += 128) {   // 4 elements per thread
    int m = e >> 4, n = e & 15;
    int b = btile + m, h = htile + n;
    size_t idx = (size_t)b * Hn + h;
    float f = fast_sigmoid(gAcc[0][m][n]);
    float i = fast_sigmoid(gAcc[1][m][n]);
    float g = fast_tanh   (gAcc[2][m][n]);
    float o = fast_sigmoid(gAcc[3][m][n]);
    float c  = f * Cst[idx] + i * g;
    float hn = o * fast_tanh(c);
    Cst[idx]   = c;
    Hnext[idx] = f2bf(hn);
    outT[idx]  = hn;
  }
}

// ---------- finalize: hx = outputs[T-1], cx = C ----------
__global__ void finalize_kernel(const float* __restrict__ lastOut,
                                const float* __restrict__ C,
                                float* __restrict__ hxOut,
                                float* __restrict__ cxOut) {
  int i = blockIdx.x * blockDim.x + threadIdx.x;
  hxOut[i] = lastOut[i];
  cxOut[i] = C[i];
}

extern "C" void kernel_launch(void* const* d_in, const int* in_sizes, int n_in,
                              void* d_out, int out_size, void* d_ws, size_t ws_size,
                              hipStream_t stream) {
  const float* inputs = (const float*)d_in[0];
  const float* Wf  = (const float*)d_in[1];
  const float* bf_ = (const float*)d_in[2];
  const float* Wi  = (const float*)d_in[3];
  const float* bi_ = (const float*)d_in[4];
  const float* Wg  = (const float*)d_in[5];
  const float* bg_ = (const float*)d_in[6];
  const float* Wo  = (const float*)d_in[7];
  const float* bo_ = (const float*)d_in[8];
  float* out = (float*)d_out;

  // workspace layout (~84.9 MB): Xbf16 | Wbf16[4] | H ping | H pong | C fp32
  unsigned short* Xbf = (unsigned short*)d_ws;
  unsigned short* Wbf = Xbf + (size_t)Tn * Bn * Dn;
  unsigned short* H0  = Wbf + (size_t)4 * Hn * Kn;
  unsigned short* H1  = H0 + (size_t)Bn * Hn;
  float*          C   = (float*)(H1 + (size_t)Bn * Hn);

  {
    long long n = (long long)Tn * Bn * Dn;
    cvt_bf16_kernel<<<(unsigned)((n + 255) / 256), 256, 0, stream>>>(inputs, Xbf, n);
  }
  {
    long long n = (long long)4 * Hn * Kn;
    cvt_weights_kernel<<<(unsigned)((n + 255) / 256), 256, 0, stream>>>(Wf, Wi, Wg, Wo, Wbf);
  }
  init_state_kernel<<<(Bn * Hn) / 256, 256, 0, stream>>>(H0, H1, C);

  dim3 grid(Hn / 16, Bn / 32); // 64 x 4 = 256 blocks
  for (int t = 0; t < Tn; ++t) {
    const unsigned short* Hin  = (t & 1) ? H1 : H0;
    unsigned short*       Hout = (t & 1) ? H0 : H1;
    lstm_step_kernel<<<grid, 128, 0, stream>>>(
        Xbf + (size_t)t * Bn * Dn, Wbf, Hin, Hout, C,
        bf_, bi_, bg_, bo_, out + (size_t)t * Bn * Hn);
  }

  finalize_kernel<<<(Bn * Hn) / 256, 256, 0, stream>>>(
      out + (size_t)(Tn - 1) * Bn * Hn, C,
      out + (size_t)Tn * Bn * Hn,
      out + (size_t)Tn * Bn * Hn + (size_t)Bn * Hn);
}
